// GeneralizedInferenceNetwork_52982716563958
// MI455X (gfx1250) — matrix operand order repaired and checked by reading, not compiled, not run on hardware
//
#include <hip/hip_runtime.h>
#include <hip/hip_bf16.h>
#include <cstdint>

// ---------------------------------------------------------------------------
// CDNA5 (gfx1250): GEMMs on v_wmma_f32_16x16x32_bf16, weight tiles staged by
// the Tensor Data Mover (tensor_load_to_lds + s_wait_tensorcnt), flash
// attention with online softmax, scalar tail kernels for the Kikuchi energy.
// ---------------------------------------------------------------------------

typedef __attribute__((ext_vector_type(16))) __bf16 v16bf;
typedef __attribute__((ext_vector_type(8)))  float  v8f;
typedef __attribute__((ext_vector_type(4)))  unsigned int v4u;
typedef __attribute__((ext_vector_type(8)))  int v8i;
typedef __attribute__((ext_vector_type(4)))  int v4i;

#if __has_builtin(__builtin_amdgcn_tensor_load_to_lds)
#define HAVE_TDM 1
#else
#define HAVE_TDM 0
#endif

union FragU {
    v16bf v;
    unsigned short u[16];
};

__device__ __forceinline__ unsigned short f2bf(float f) {
    unsigned u = __float_as_uint(f);
    unsigned r = u + 0x7fffu + ((u >> 16) & 1u);   // round-to-nearest-even
    return (unsigned short)(r >> 16);
}

__device__ __forceinline__ v8f wmma_bf16(v16bf a, v16bf b, v8f c) {
    return __builtin_amdgcn_wmma_f32_16x16x32_bf16(
        /*neg_a=*/false, a, /*neg_b=*/false, b,
        /*c_mod=*/(short)0, c, /*reuse_a=*/false, /*reuse_b=*/false);
}

// A-fragment (16x32 bf16): M across lanes, K across VGPRs, contiguous
// 16-K per lane half (k = 16h + s). `lds` = tile origin, row stride in elements.
__device__ __forceinline__ v16bf load_a_frag(const unsigned short* lds,
                                             int stride, int lane) {
    int row = lane & 15;
    int kh  = (lane >> 4) * 16;
    const unsigned short* p = lds + row * stride;
    FragU f;
#pragma unroll
    for (int j = 0; j < 4; ++j) {
        int k = kh + 2 * j;
        f.u[2 * j]     = p[k];
        f.u[2 * j + 1] = p[k + 1];
    }
#pragma unroll
    for (int j = 0; j < 4; ++j) {
        int k = kh + 8 + 2 * j;
        f.u[8 + 2 * j] = p[k];
        f.u[9 + 2 * j] = p[k + 1];
    }
    return f.v;
}

// B-fragment from an N-major LDS tile: ldsT[n][k] (attention path).
__device__ __forceinline__ v16bf load_b_frag(const unsigned short* ldsT,
                                             int stride, int lane) {
    int col = lane & 15;
    int kh  = (lane >> 4) * 16;
    const unsigned short* p = ldsT + col * stride + kh;
    FragU f;
#pragma unroll
    for (int j = 0; j < 16; ++j) f.u[j] = p[j];
    return f.v;
}

// B-fragment from a K-major LDS tile (TDM-delivered): lds[k][n], ncols wide.
__device__ __forceinline__ v16bf load_b_frag_kmaj(const unsigned short* lds,
                                                  int ncols, int col, int lane) {
    int kh = (lane >> 4) * 16;
    int n  = col + (lane & 15);
    FragU f;
#pragma unroll
    for (int j = 0; j < 16; ++j) f.u[j] = lds[(kh + j) * ncols + n];
    return f.v;
}

// ---------------------------------------------------------------------------
// TDM: load a 32xBN bf16 tile of W[K,N] (row-major, bf16) into LDS.
// D# per ISA 08_async_tensor §8.3/8.4: count=1, type=2 (image),
// data_size=1 (2 bytes), 2D tile, dim0 stride = N elements.
// ---------------------------------------------------------------------------
__device__ __forceinline__ void tdm_issue(unsigned short* ldsDst,
                                          const unsigned short* gsrc,
                                          int N, int K, int k0, int bn) {
#if HAVE_TDM
    unsigned long long ga =
        (unsigned long long)(uintptr_t)(gsrc + (size_t)k0 * N + bn);
    unsigned lds_off = (unsigned)(uintptr_t)ldsDst;   // low 32 bits = LDS offset
    v4u g0;
    g0[0] = 1u;                                            // count=1 (valid D#)
    g0[1] = lds_off;                                       // lds_addr (bytes)
    g0[2] = (unsigned)ga;                                  // global_addr[31:0]
    g0[3] = (unsigned)((ga >> 32) & 0x01FFFFFFull) | (2u << 30);  // addr hi | type=2
    v8i g1;
    g1[0] = (int)(1u << 16);                               // data_size=1 -> 2B
    g1[1] = (int)(((unsigned)N & 0xFFFFu) << 16);          // tensor_dim0 lo16
    g1[2] = (int)(((unsigned)N >> 16) |
                  (((unsigned)K & 0xFFFFu) << 16));        // dim0 hi / dim1 lo
    g1[3] = (int)(((unsigned)K >> 16) | (64u << 16));      // dim1 hi / tile_dim0=64
    g1[4] = (int)32u;                                      // tile_dim1=32, tile_dim2=0
    g1[5] = (int)(unsigned)N;                              // tensor_dim0_stride lo32
    g1[6] = 0;
    g1[7] = 0;
    v4i z4 = {0, 0, 0, 0};
#if defined(__clang_major__) && __clang_major__ >= 23
    v8i z8 = {0, 0, 0, 0, 0, 0, 0, 0};
    __builtin_amdgcn_tensor_load_to_lds(g0, g1, z4, z4, z8, 0);
#else
    __builtin_amdgcn_tensor_load_to_lds(g0, g1, z4, z4, 0);
#endif
#else
    (void)ldsDst; (void)gsrc; (void)N; (void)K; (void)k0; (void)bn;
#endif
}

// ---------------------------------------------------------------------------
// GEMM: C[M,N] = act(A[M,K](f32) @ Wbf[K,N](bf16) + bias) (+ residual)
// Block 128 threads = 4 waves, tile 64x64, BK=32; W tiles via TDM with
// double-buffered LDS; A tile staged f32->bf16 by VALU (activations are f32).
// ---------------------------------------------------------------------------
template <int ACT, int RES>
__global__ __launch_bounds__(128)
void gemm_wmma(const float* __restrict__ A, const unsigned short* __restrict__ Wbf,
               const float* __restrict__ bias, const float* __restrict__ res,
               float* __restrict__ C, int M, int N, int K) {
    constexpr int BM = 64, BN = 64, BK = 32;
    constexpr int AS = 36;
    __shared__ unsigned short As[BM * AS];
    __shared__ unsigned short Bs[2][BK * BN];    // k-major, 4 KB each

    int tid  = threadIdx.x;
    int lane = tid & 31;
    int wave = tid >> 5;
    int wm   = (wave >> 1) * 32;
    int wn   = (wave & 1) * 32;
    int bm   = blockIdx.y * BM;
    int bn   = blockIdx.x * BN;

    v8f acc[2][2] = {};

#if HAVE_TDM
    if (wave == 0) tdm_issue(&Bs[0][0], Wbf, N, K, 0, bn);   // prologue tile
#endif

    int pb = 0;
    for (int k0 = 0; k0 < K; k0 += BK, pb ^= 1) {
        // Stage A tile 64x32 (f32 -> bf16), coalesced along K.
#pragma unroll
        for (int i = 0; i < 16; ++i) {
            int idx = tid + i * 128;
            int r = idx >> 5, c = idx & 31;
            int gm = bm + r, gk = k0 + c;
            float v = (gm < M) ? A[(size_t)gm * K + gk] : 0.f;
            As[r * AS + c] = f2bf(v);
        }
        if (k0 + BK < K) {   // prefetch next A k-slab (global_prefetch_b8)
            int gm = bm + (tid >> 1);
            if (gm < M) __builtin_prefetch(&A[(size_t)gm * K + k0 + BK], 0, 3);
        }

#if HAVE_TDM
        if (wave == 0) __builtin_amdgcn_s_wait_tensorcnt(0); // tile pb arrived
        __syncthreads();
        if (wave == 0 && k0 + BK < K)
            tdm_issue(&Bs[pb ^ 1][0], Wbf, N, K, k0 + BK, bn);
#else
        // Fallback: cooperative bf16 copy of the 32x64 W tile.
#pragma unroll
        for (int i = 0; i < 16; ++i) {
            int idx = tid + i * 128;
            int kk = idx >> 6, nn = idx & 63;
            Bs[pb][kk * BN + nn] = Wbf[(size_t)(k0 + kk) * N + bn + nn];
        }
        __syncthreads();
#endif

        v16bf afr[2], bfr[2];
        afr[0] = load_a_frag(&As[(wm + 0)  * AS], AS, lane);
        afr[1] = load_a_frag(&As[(wm + 16) * AS], AS, lane);
        bfr[0] = load_b_frag_kmaj(&Bs[pb][0], BN, wn + 0,  lane);
        bfr[1] = load_b_frag_kmaj(&Bs[pb][0], BN, wn + 16, lane);
#pragma unroll
        for (int mi = 0; mi < 2; ++mi)
#pragma unroll
            for (int ni = 0; ni < 2; ++ni)
                acc[mi][ni] = wmma_bf16(afr[mi], bfr[ni], acc[mi][ni]);
        __syncthreads();
    }

    // C/D layout: VGPR r -> M = r + 8*(lane>=16); N = lane&15.
#pragma unroll
    for (int mi = 0; mi < 2; ++mi) {
#pragma unroll
        for (int ni = 0; ni < 2; ++ni) {
            int col  = bn + wn + ni * 16 + (lane & 15);
            int row0 = bm + wm + mi * 16 + ((lane >> 4) << 3);
#pragma unroll
            for (int r = 0; r < 8; ++r) {
                int row = row0 + r;
                if (row < M && col < N) {
                    float v = acc[mi][ni][r] + bias[col];
                    if (ACT) v = fmaxf(v, 0.f);
                    if (RES) v += res[(size_t)row * N + col];
                    C[(size_t)row * N + col] = v;
                }
            }
        }
    }
}

// f32 -> bf16 bulk conversion (weights, once per launch).
__global__ void f32_to_bf16(const float* __restrict__ src,
                            unsigned short* __restrict__ dst, int count) {
    int i = blockIdx.x * blockDim.x + threadIdx.x;
    if (i < count) dst[i] = f2bf(src[i]);
}

// ---------------------------------------------------------------------------
// Flash attention (no scale, per reference): O = softmax(Q K^T) V.
// Block = 64 threads (2 waves), 32 query rows / block, 32 keys per tile.
// ---------------------------------------------------------------------------
__global__ __launch_bounds__(64)
void attn_flash(const float* __restrict__ Q, const float* __restrict__ Km,
                const float* __restrict__ V, float* __restrict__ O, int Nseq) {
    constexpr int D_ = 256;
    constexpr int QROWS = 32;
    constexpr int KT = 32;
    constexpr int QS = 264;
    constexpr int KS = 264;
    constexpr int VS = 36;
    __shared__ unsigned short Qs[QROWS * QS];
    __shared__ unsigned short Ks[KT * KS];
    __shared__ unsigned short VsT[D_ * VS];
    __shared__ unsigned short Pws[2][16 * 32];

    int tid = threadIdx.x, lane = tid & 31, wave = tid >> 5;
    int qbase = blockIdx.x * QROWS;
    int wq = wave * 16;

    for (int i = tid; i < QROWS * D_; i += 64) {
        int r = i >> 8, c = i & 255;
        Qs[r * QS + c] = f2bf(Q[(size_t)(qbase + r) * D_ + c]);
    }
    __syncthreads();

    v8f o_acc[16] = {};
    float m_s[8], l_s[8];
#pragma unroll
    for (int r = 0; r < 8; ++r) { m_s[r] = -1e30f; l_s[r] = 0.f; }

    for (int kt = 0; kt < Nseq; kt += KT) {
        for (int i = tid; i < KT * D_; i += 64) {
            int r = i >> 8, c = i & 255;
            Ks[r * KS + c]  = f2bf(Km[(size_t)(kt + r) * D_ + c]);
            VsT[c * VS + r] = f2bf(V[(size_t)(kt + r) * D_ + c]);
        }
        __syncthreads();

        v8f s0 = {}, s1 = {};
#pragma unroll
        for (int kk = 0; kk < D_; kk += 32) {
            v16bf a  = load_a_frag(&Qs[wq * QS + kk], QS, lane);
            v16bf b0 = load_b_frag(&Ks[0  * KS + kk], KS, lane);
            v16bf b1 = load_b_frag(&Ks[16 * KS + kk], KS, lane);
            s0 = wmma_bf16(a, b0, s0);
            s1 = wmma_bf16(a, b1, s1);
        }

#pragma unroll
        for (int r = 0; r < 8; ++r) {
            float mx = fmaxf(s0[r], s1[r]);
            mx = fmaxf(mx, __shfl_xor(mx, 1, 32));
            mx = fmaxf(mx, __shfl_xor(mx, 2, 32));
            mx = fmaxf(mx, __shfl_xor(mx, 4, 32));
            mx = fmaxf(mx, __shfl_xor(mx, 8, 32));
            float mnew  = fmaxf(m_s[r], mx);
            float scale = __expf(m_s[r] - mnew);
            float p0 = __expf(s0[r] - mnew);
            float p1 = __expf(s1[r] - mnew);
            s0[r] = p0; s1[r] = p1;
            float rs = p0 + p1;
            rs += __shfl_xor(rs, 1, 32);
            rs += __shfl_xor(rs, 2, 32);
            rs += __shfl_xor(rs, 4, 32);
            rs += __shfl_xor(rs, 8, 32);
            l_s[r] = l_s[r] * scale + rs;
            m_s[r] = mnew;
#pragma unroll
            for (int t = 0; t < 16; ++t) o_acc[t][r] *= scale;
        }

        unsigned short* pw = &Pws[wave][0];
        int prow0 = (lane >> 4) << 3;
        int pcol  = lane & 15;
#pragma unroll
        for (int r = 0; r < 8; ++r) {
            pw[(prow0 + r) * 32 + pcol]      = f2bf(s0[r]);
            pw[(prow0 + r) * 32 + 16 + pcol] = f2bf(s1[r]);
        }
        __syncthreads();

        v16bf pa = load_a_frag(pw, 32, lane);
#pragma unroll
        for (int t = 0; t < 16; ++t) {
            v16bf bv = load_b_frag(&VsT[(t * 16) * VS], VS, lane);
            o_acc[t] = wmma_bf16(pa, bv, o_acc[t]);
        }
        __syncthreads();
    }

#pragma unroll
    for (int t = 0; t < 16; ++t) {
        int col  = t * 16 + (lane & 15);
        int row0 = qbase + wq + ((lane >> 4) << 3);
#pragma unroll
        for (int r = 0; r < 8; ++r)
            O[(size_t)(row0 + r) * D_ + col] = o_acc[t][r] / l_s[r];
    }
}

// x = LayerNorm(d + x) * g + b  (one row of 256 per block of 256 threads)
__global__ __launch_bounds__(256)
void add_ln(const float* __restrict__ dsum, float* __restrict__ x,
            const float* __restrict__ g, const float* __restrict__ b) {
    constexpr int D_ = 256;
    __shared__ float red[256];
    int row = blockIdx.x, t = threadIdx.x;
    float v = dsum[(size_t)row * D_ + t] + x[(size_t)row * D_ + t];
    red[t] = v; __syncthreads();
    for (int s = 128; s > 0; s >>= 1) { if (t < s) red[t] += red[t + s]; __syncthreads(); }
    float mu = red[0] * (1.f / D_); __syncthreads();
    float d = v - mu;
    red[t] = d * d; __syncthreads();
    for (int s = 128; s > 0; s >>= 1) { if (t < s) red[t] += red[t + s]; __syncthreads(); }
    float var = red[0] * (1.f / D_);
    x[(size_t)row * D_ + t] = d * rsqrtf(var + 1e-5f) * g[t] + b[t];
}

// feats[r, j*D+d] = x[idx[r,j]*D + d]
__global__ void gather_feats(const float* __restrict__ x, const int* __restrict__ idx,
                             float* __restrict__ feats, int R0) {
    constexpr int D_ = 256;
    int i = blockIdx.x * blockDim.x + threadIdx.x;
    if (i >= R0 * 4 * D_) return;
    int d = i & 255;
    int j = (i >> 8) & 3;
    int r = i >> 10;
    feats[(size_t)r * 1024 + j * 256 + d] = x[(size_t)idx[r * 4 + j] * D_ + d];
}

// Output head: logits -> softmax -> beliefs; energy(+1) of beliefs.
__global__ __launch_bounds__(128)
void beliefs_energy(const float* __restrict__ h, const float* __restrict__ Wout,
                    const float* __restrict__ bout, const float* __restrict__ lphi,
                    float* __restrict__ beliefs, float* __restrict__ out,
                    int R0, int K) {
    __shared__ float red[128];
    int t = threadIdx.x;
    int row = blockIdx.x * 8 + (t >> 4);
    int o = t & 15;
    float epart = 0.f;
    if (row < R0) {
        float acc = bout[o];
        const float* hr = h + (size_t)row * K;
        for (int k = 0; k < K; ++k) acc += hr[k] * Wout[k * 16 + o];
        float mx = acc;
        mx = fmaxf(mx, __shfl_xor(mx, 1, 32));
        mx = fmaxf(mx, __shfl_xor(mx, 2, 32));
        mx = fmaxf(mx, __shfl_xor(mx, 4, 32));
        mx = fmaxf(mx, __shfl_xor(mx, 8, 32));
        float ex = __expf(acc - mx);
        float s = ex;
        s += __shfl_xor(s, 1, 32); s += __shfl_xor(s, 2, 32);
        s += __shfl_xor(s, 4, 32); s += __shfl_xor(s, 8, 32);
        float bv = ex / s;
        beliefs[(size_t)row * 16 + o] = bv;
        float bc = fmaxf(bv, 1e-8f);
        epart = bc * (__logf(bc) - lphi[(size_t)row * 16 + o]);
    }
    red[t] = epart; __syncthreads();
    for (int s = 64; s > 0; s >>= 1) { if (t < s) red[t] += red[t + s]; __syncthreads(); }
    if (t == 0) atomicAdd(&out[0], red[0]);
}

// Horizontal interior edges: h_bel + mism + energy(-1).
__global__ __launch_bounds__(256)
void hedge_kernel(const float* __restrict__ bel, const float* __restrict__ lphi,
                  float* __restrict__ hbel, float* __restrict__ out, int n) {
    __shared__ float redE[256], redM[256];
    int m = n - 1, Eh = (n - 2) * m;
    int t = threadIdx.x, e = blockIdx.x * 256 + t;
    float epart = 0.f, mpart = 0.f;
    if (e < Eh) {
        int r = e / m + 1, c = e % m;
        const float* ba = bel + (size_t)((r - 1) * m + c) * 16;
        const float* bb = bel + (size_t)(r * m + c) * 16;
        float av[4], bw[4];
#pragma unroll
        for (int s2 = 0; s2 < 2; ++s2)
            for (int s3 = 0; s3 < 2; ++s3) {
                float a = 0.f;
                for (int s0 = 0; s0 < 2; ++s0)
                    for (int s1 = 0; s1 < 2; ++s1) a += ba[s0 * 8 + s1 * 4 + s2 * 2 + s3];
                av[s2 * 2 + s3] = a;
            }
#pragma unroll
        for (int s0 = 0; s0 < 2; ++s0)
            for (int s1 = 0; s1 < 2; ++s1) {
                float a = 0.f;
                for (int s2 = 0; s2 < 2; ++s2)
                    for (int s3 = 0; s3 < 2; ++s3) a += bb[s0 * 8 + s1 * 4 + s2 * 2 + s3];
                bw[s0 * 2 + s1] = a;
            }
#pragma unroll
        for (int j = 0; j < 4; ++j) {
            float hm = 0.5f * (av[j] + bw[j]);
            hbel[(size_t)e * 4 + j] = hm;
            float da = av[j] - hm, db = bw[j] - hm;
            mpart += da * da + db * db;
            float bc = fmaxf(hm, 1e-8f);
            epart -= bc * (__logf(bc) - lphi[(size_t)e * 4 + j]);
        }
    }
    redE[t] = epart; redM[t] = mpart; __syncthreads();
    for (int s = 128; s > 0; s >>= 1) {
        if (t < s) { redE[t] += redE[t + s]; redM[t] += redM[t + s]; }
        __syncthreads();
    }
    if (t == 0) { atomicAdd(&out[0], redE[0]); atomicAdd(&out[1], redM[0]); }
}

// Vertical interior edges: v_bel + mism + energy(-1).
__global__ __launch_bounds__(256)
void vedge_kernel(const float* __restrict__ bel, const float* __restrict__ lphi,
                  float* __restrict__ vbel, float* __restrict__ out, int n) {
    __shared__ float redE[256], redM[256];
    int m = n - 1, Ev = m * (n - 2);
    int t = threadIdx.x, e = blockIdx.x * 256 + t;
    float epart = 0.f, mpart = 0.f;
    if (e < Ev) {
        int r = e / (n - 2), c = e % (n - 2) + 1;
        const float* bl = bel + (size_t)(r * m + (c - 1)) * 16;
        const float* br = bel + (size_t)(r * m + c) * 16;
        float lv[4], rv[4];
#pragma unroll
        for (int s1 = 0; s1 < 2; ++s1)
            for (int s3 = 0; s3 < 2; ++s3) {
                float a = 0.f;
                for (int s0 = 0; s0 < 2; ++s0)
                    for (int s2 = 0; s2 < 2; ++s2) a += bl[s0 * 8 + s1 * 4 + s2 * 2 + s3];
                lv[s1 * 2 + s3] = a;
            }
#pragma unroll
        for (int s0 = 0; s0 < 2; ++s0)
            for (int s2 = 0; s2 < 2; ++s2) {
                float a = 0.f;
                for (int s1 = 0; s1 < 2; ++s1)
                    for (int s3 = 0; s3 < 2; ++s3) a += br[s0 * 8 + s1 * 4 + s2 * 2 + s3];
                rv[s0 * 2 + s2] = a;
            }
#pragma unroll
        for (int j = 0; j < 4; ++j) {
            float vm = 0.5f * (lv[j] + rv[j]);
            vbel[(size_t)e * 4 + j] = vm;
            float da = lv[j] - vm, db = rv[j] - vm;
            mpart += da * da + db * db;
            float bc = fmaxf(vm, 1e-8f);
            epart -= bc * (__logf(bc) - lphi[(size_t)e * 4 + j]);
        }
    }
    redE[t] = epart; redM[t] = mpart; __syncthreads();
    for (int s = 128; s > 0; s >>= 1) {
        if (t < s) { redE[t] += redE[t + s]; redM[t] += redM[t + s]; }
        __syncthreads();
    }
    if (t == 0) { atomicAdd(&out[0], redE[0]); atomicAdd(&out[1], redM[0]); }
}

// Interior nodes: n_bel + mism + energy(+1).
__global__ __launch_bounds__(256)
void node_kernel(const float* __restrict__ hbel, const float* __restrict__ vbel,
                 const float* __restrict__ lphi, float* __restrict__ out, int n) {
    __shared__ float redE[256], redM[256];
    int m = n - 1, Vn = (n - 2) * (n - 2);
    int t = threadIdx.x, e = blockIdx.x * 256 + t;
    float epart = 0.f, mpart = 0.f;
    if (e < Vn) {
        int r = e / (n - 2) + 1, c = e % (n - 2) + 1;
        const float* hl = hbel + (size_t)((r - 1) * m + (c - 1)) * 4;
        const float* hr = hbel + (size_t)((r - 1) * m + c) * 4;
        const float* vu = vbel + (size_t)((r - 1) * (n - 2) + (c - 1)) * 4;
        const float* vd = vbel + (size_t)(r * (n - 2) + (c - 1)) * 4;
        float st[4][2];
        st[0][0] = hl[0] + hl[2]; st[0][1] = hl[1] + hl[3];
        st[1][0] = hr[0] + hr[1]; st[1][1] = hr[2] + hr[3];
        st[2][0] = vu[0] + vu[2]; st[2][1] = vu[1] + vu[3];
        st[3][0] = vd[0] + vd[1]; st[3][1] = vd[2] + vd[3];
#pragma unroll
        for (int j = 0; j < 2; ++j) {
            float nb = 0.25f * (st[0][j] + st[1][j] + st[2][j] + st[3][j]);
            for (int q = 0; q < 4; ++q) {
                float d = st[q][j] - nb;
                mpart += d * d;
            }
            float bc = fmaxf(nb, 1e-8f);
            epart += bc * (__logf(bc) - lphi[(size_t)e * 2 + j]);
        }
    }
    redE[t] = epart; redM[t] = mpart; __syncthreads();
    for (int s = 128; s > 0; s >>= 1) {
        if (t < s) { redE[t] += redE[t + s]; redM[t] += redM[t + s]; }
        __syncthreads();
    }
    if (t == 0) { atomicAdd(&out[0], redE[0]); atomicAdd(&out[1], redM[0]); }
}

// ---------------------------------------------------------------------------
// Host orchestration (graph-capture safe: only async ops on `stream`).
// ---------------------------------------------------------------------------
extern "C" void kernel_launch(void* const* d_in, const int* in_sizes, int n_in,
                              void* d_out, int out_size, void* d_ws, size_t ws_size,
                              hipStream_t stream) {
    (void)in_sizes; (void)n_in; (void)out_size; (void)ws_size;
    constexpr int n = 64, D = 256, L = 2, HID = 1024;
    constexpr int Nseq = n * n;
    constexpr int m = n - 1;
    constexpr int R0 = m * m;
    constexpr int Eh = (n - 2) * m;
    constexpr int Ev = m * (n - 2);
    constexpr int Vn = (n - 2) * (n - 2);
    constexpr int AW = L * D * D;        // attn weight elems per tensor
    constexpr int MW = HID * HID;        // mlp weight elems per tensor

    const float* node_emb = (const float*)d_in[0];
    const float* Wq = (const float*)d_in[1];  const float* bq = (const float*)d_in[2];
    const float* Wk = (const float*)d_in[3];  const float* bk = (const float*)d_in[4];
    const float* Wv = (const float*)d_in[5];  const float* bv = (const float*)d_in[6];
    const float* Wd = (const float*)d_in[7];  const float* bd = (const float*)d_in[8];
    const float* lng = (const float*)d_in[9]; const float* lnb = (const float*)d_in[10];
    const float* w11 = (const float*)d_in[11]; const float* b11 = (const float*)d_in[12];
    const float* w12 = (const float*)d_in[13]; const float* b12 = (const float*)d_in[14];
    const float* w21 = (const float*)d_in[15]; const float* b21 = (const float*)d_in[16];
    const float* w22 = (const float*)d_in[17]; const float* b22 = (const float*)d_in[18];
    const float* outw = (const float*)d_in[19]; const float* outb = (const float*)d_in[20];
    const float* lphi0 = (const float*)d_in[21];
    const float* lphih = (const float*)d_in[22];
    const float* lphiv = (const float*)d_in[23];
    const float* lphin = (const float*)d_in[24];
    const int* r0_idx = (const int*)d_in[25];

    // Workspace layout (floats), t1 aliases q..dp (dead after attention).
    float* ws = (float*)d_ws;
    constexpr size_t SEQ = (size_t)Nseq * D;        // 1,048,576
    constexpr size_t MLP = (size_t)R0 * HID;        // 4,064,256
    float* x     = ws;
    float* q     = x + SEQ;
    float* kbuf  = q + SEQ;
    float* vbuf  = kbuf + SEQ;
    float* ctx   = vbuf + SEQ;
    float* dp    = ctx + SEQ;
    float* feats = dp + SEQ;
    float* t2    = feats + MLP;
    float* bel   = t2 + MLP;
    float* hb    = bel + (size_t)R0 * 16;
    float* vb    = hb + (size_t)Eh * 4;
    float* t1    = q;  // alias: spans q..dp (5*SEQ floats >= MLP)

    // bf16 weight copies.
    unsigned short* wqb  = (unsigned short*)(vb + (size_t)Ev * 4);
    unsigned short* wkb  = wqb + AW;
    unsigned short* wvb  = wkb + AW;
    unsigned short* wdb  = wvb + AW;
    unsigned short* w11b = wdb + AW;
    unsigned short* w12b = w11b + MW;
    unsigned short* w21b = w12b + MW;
    unsigned short* w22b = w21b + MW;

    float* out = (float*)d_out;

    // One-time (per launch) weight conversion f32 -> bf16.
    f32_to_bf16<<<(AW + 255) / 256, 256, 0, stream>>>(Wq, wqb, AW);
    f32_to_bf16<<<(AW + 255) / 256, 256, 0, stream>>>(Wk, wkb, AW);
    f32_to_bf16<<<(AW + 255) / 256, 256, 0, stream>>>(Wv, wvb, AW);
    f32_to_bf16<<<(AW + 255) / 256, 256, 0, stream>>>(Wd, wdb, AW);
    f32_to_bf16<<<(MW + 255) / 256, 256, 0, stream>>>(w11, w11b, MW);
    f32_to_bf16<<<(MW + 255) / 256, 256, 0, stream>>>(w12, w12b, MW);
    f32_to_bf16<<<(MW + 255) / 256, 256, 0, stream>>>(w21, w21b, MW);
    f32_to_bf16<<<(MW + 255) / 256, 256, 0, stream>>>(w22, w22b, MW);

    hipMemcpyAsync(x, node_emb, SEQ * sizeof(float), hipMemcpyDeviceToDevice, stream);

    dim3 gproj(D / 64, Nseq / 64);
    for (int l = 0; l < L; ++l) {
        const unsigned short* Wql = wqb + (size_t)l * D * D;
        const unsigned short* Wkl = wkb + (size_t)l * D * D;
        const unsigned short* Wvl = wvb + (size_t)l * D * D;
        const unsigned short* Wdl = wdb + (size_t)l * D * D;
        const float* bql = bq + (size_t)l * D;
        const float* bkl = bk + (size_t)l * D;
        const float* bvl = bv + (size_t)l * D;
        const float* bdl = bd + (size_t)l * D;
        gemm_wmma<0, 0><<<gproj, 128, 0, stream>>>(x, Wql, bql, nullptr, q,    Nseq, D, D);
        gemm_wmma<0, 0><<<gproj, 128, 0, stream>>>(x, Wkl, bkl, nullptr, kbuf, Nseq, D, D);
        gemm_wmma<0, 0><<<gproj, 128, 0, stream>>>(x, Wvl, bvl, nullptr, vbuf, Nseq, D, D);
        attn_flash<<<Nseq / 32, 64, 0, stream>>>(q, kbuf, vbuf, ctx, Nseq);
        gemm_wmma<0, 0><<<gproj, 128, 0, stream>>>(ctx, Wdl, bdl, nullptr, dp, Nseq, D, D);
        add_ln<<<Nseq, 256, 0, stream>>>(dp, x, lng + (size_t)l * D, lnb + (size_t)l * D);
    }

    gather_feats<<<(R0 * 4 * D + 255) / 256, 256, 0, stream>>>(x, r0_idx, feats, R0);

    dim3 gmlp(HID / 64, (R0 + 63) / 64);
    gemm_wmma<1, 0><<<gmlp, 128, 0, stream>>>(feats, w11b, b11, nullptr, t1, R0, HID, HID);
    gemm_wmma<1, 1><<<gmlp, 128, 0, stream>>>(t1, w12b, b12, feats, t2, R0, HID, HID);
    gemm_wmma<1, 0><<<gmlp, 128, 0, stream>>>(t2, w21b, b21, nullptr, t1, R0, HID, HID);
    gemm_wmma<1, 1><<<gmlp, 128, 0, stream>>>(t1, w22b, b22, t2, feats, R0, HID, HID);

    hipMemsetAsync(d_out, 0, 2 * sizeof(float), stream);
    beliefs_energy<<<(R0 + 7) / 8, 128, 0, stream>>>(feats, outw, outb, lphi0, bel, out, R0, HID);
    hedge_kernel<<<(Eh + 255) / 256, 256, 0, stream>>>(bel, lphih, hb, out, n);
    vedge_kernel<<<(Ev + 255) / 256, 256, 0, stream>>>(bel, lphiv, vb, out, n);
    node_kernel<<<(Vn + 255) / 256, 256, 0, stream>>>(hb, vb, lphin, out, n);
}
